// TransformerCovarianceModule_20444044329463
// MI455X (gfx1250) — matrix operand order repaired and checked by reading, not compiled, not run on hardware
//
#include <hip/hip_runtime.h>
#include <hip/hip_bf16.h>

// ---------------------------------------------------------------------------
// Types for CDNA5 WMMA (wave32)
// ---------------------------------------------------------------------------
typedef __bf16 bf16_t;
typedef __attribute__((ext_vector_type(16))) __bf16 v16bf;
typedef __attribute__((ext_vector_type(8)))  __bf16 v8bf;
typedef __attribute__((ext_vector_type(4)))  __bf16 v4bf;
typedef __attribute__((ext_vector_type(8)))  float  v8f;

#define C_IN   256
#define N_PIX  4096
#define NH     8
#define HD     32
#define MCH    64          // attention m-chunk

__device__ __forceinline__ v8f wmma_bf16(v16bf a, v16bf b, v8f c) {
    // D = A(16x32 bf16) * B(32x16 bf16) + C(16x16 f32)
    return __builtin_amdgcn_wmma_f32_16x16x32_bf16(
        false, a, false, b, (short)0, c, false, false);
}

// Build a v16bf A-operand from two contiguous 8-element (16B) chunks.
__device__ __forceinline__ v16bf ld2(const bf16_t* p0, const bf16_t* p1) {
    v8bf lo = *(const v8bf*)p0;
    v8bf hi = *(const v8bf*)p1;
    v16bf r;
#pragma unroll
    for (int i = 0; i < 8; ++i) { r[i] = lo[i]; r[i + 8] = hi[i]; }
    return r;
}

// Cross-lane reductions within each 16-lane half (WMMA C-tile rows are striped
// across lanes 0-15 / 16-31; xor masks 1,2,4,8 stay inside a half).
__device__ __forceinline__ float hmax16(float v) {
#pragma unroll
    for (int m = 1; m <= 8; m <<= 1) v = fmaxf(v, __shfl_xor(v, m, 32));
    return v;
}
__device__ __forceinline__ float hsum16(float v) {
#pragma unroll
    for (int m = 1; m <= 8; m <<= 1) v += __shfl_xor(v, m, 32);
    return v;
}

// LDS byte offset of a __shared__ object (generic-pointer low 32 bits).
__device__ __forceinline__ unsigned ldsoff(const void* p) {
    return (unsigned)(uintptr_t)p;
}

// ---------------------------------------------------------------------------
// Kernel 0: one-time fp32 -> bf16 conversion of both weight matrices.
// ---------------------------------------------------------------------------
__global__ __launch_bounds__(256) void wconv_kernel(
    const float* __restrict__ wq, const float* __restrict__ wp,
    bf16_t* __restrict__ wqb, bf16_t* __restrict__ wpb)
{
    const int i = blockIdx.x * 256 + threadIdx.x;
    if (i < 768 * C_IN) wqb[i] = (bf16_t)wq[i];
    if (i < C_IN * C_IN) wpb[i] = (bf16_t)wp[i];
}

// ---------------------------------------------------------------------------
// Kernel 1: QKV projection -> bf16 Q/K/V in WMMA-friendly layouts
//   Q,K : [b][h][n][d]          (d contiguous; Q/K stores are 16B packed)
//   V   : [b][h][d][n']         (n' = mu-permuted within 64-blocks:
//                                n' = (n&~63) | ((n&15)*4 + ((n>>4)&3)))
// ---------------------------------------------------------------------------
__global__ __launch_bounds__(256) void qkv_kernel(
    const float* __restrict__ x, const bf16_t* __restrict__ wq,
    const float* __restrict__ bq,
    bf16_t* __restrict__ Qb, bf16_t* __restrict__ Kb, bf16_t* __restrict__ Vb)
{
    __shared__ __attribute__((aligned(64))) bf16_t xt[16][32];

    const int blk = blockIdx.x;
    const int b  = blk >> 8;
    const int n0 = (blk & 255) << 4;
    const int tid = threadIdx.x;
    const int lane = tid & 31;
    const int wv = tid >> 5;
    const int lhalf = lane & 15;
    const bool hi = lane >= 16;

    // x staging assignment: one float2 per thread per k-step
    const int cs = tid >> 3;           // 0..31
    const int ns = (tid & 7) << 1;     // 0,2,..,14

    v8f acc[6];
#pragma unroll
    for (int t = 0; t < 6; ++t)
#pragma unroll
        for (int i = 0; i < 8; ++i) acc[t][i] = 0.0f;

    for (int k0 = 0; k0 < C_IN; k0 += 32) {
        __syncthreads();
        {
            const float2 xv = *(const float2*)
                &x[((size_t)b * C_IN + (k0 + cs)) * N_PIX + n0 + ns];
            xt[ns][cs]     = (bf16_t)xv.x;
            xt[ns + 1][cs] = (bf16_t)xv.y;
        }
        __syncthreads();

        v16bf bt = ld2(&xt[lhalf][hi ? 8 : 0], &xt[lhalf][hi ? 24 : 16]);
        const int cA0 = k0 + (hi ? 8 : 0);
#pragma unroll
        for (int t = 0; t < 6; ++t) {
            const int o0 = (wv + 8 * t) << 4;
            const bf16_t* wrow = wq + (size_t)(o0 + lhalf) * C_IN;
            v16bf at = ld2(wrow + cA0, wrow + cA0 + 16);
            acc[t] = wmma_bf16(at, bt, acc[t]);
        }
    }

#pragma unroll
    for (int t = 0; t < 6; ++t) {
        const int o0  = (wv + 8 * t) << 4;
        const int sec = o0 >> 8;      // 0=Q 1=K 2=V
        const int oc  = o0 & 255;
        const int h   = oc >> 5;
        const int d0  = oc & 31;
        const int n   = n0 + lhalf;
        const size_t bh = (size_t)(b * NH + h);
        if (sec < 2) {
            // lane's 8 rows -> 8 consecutive d at fixed n: one 16B store
            bf16_t* dst = (sec == 0) ? Qb : Kb;
            v8bf pk;
#pragma unroll
            for (int r = 0; r < 8; ++r)
                pk[r] = (bf16_t)(acc[t][r] + bq[o0 + r + (hi ? 8 : 0)]);
            *(v8bf*)&dst[(bh * N_PIX + n) * HD + d0 + (hi ? 8 : 0)] = pk;
        } else {
            // V: mu-permuted column within the 64-block
            const int nper = (n & ~63) | (((n & 15) << 2) | ((n >> 4) & 3));
#pragma unroll
            for (int r = 0; r < 8; ++r) {
                const int M = r + (hi ? 8 : 0);
                const float val = acc[t][r] + bq[o0 + M];
                Vb[(bh * HD + (d0 + M)) * (size_t)N_PIX + nper] = (bf16_t)val;
            }
        }
    }
}

// ---------------------------------------------------------------------------
// Kernel 2: fused flash attention.
// Workgroup = 8 waves sharing one (b,h); each wave owns 16 query rows.
// K/V tiles for each 64-wide m-chunk staged into LDS with
// global_load_async_to_lds_b128 (double-buffered, ASYNCcnt-tracked).
// P tile written to LDS in mu-order -> packed ds_store_b64.
// ---------------------------------------------------------------------------
__global__ __launch_bounds__(256) void attn_kernel(
    const bf16_t* __restrict__ Qb, const bf16_t* __restrict__ Kb,
    const bf16_t* __restrict__ Vb, bf16_t* __restrict__ AO)
{
    __shared__ __attribute__((aligned(64))) bf16_t Kl[2][MCH][HD];   // 8 KB
    __shared__ __attribute__((aligned(64))) bf16_t Vl[2][HD][MCH];   // 8 KB
    __shared__ __attribute__((aligned(64))) bf16_t pl[8][16][MCH];   // 16 KB

    const int blk  = blockIdx.x;
    const int tile = blk & 31;
    const int h    = (blk >> 5) & 7;
    const int b    = blk >> 8;
    const int tid = threadIdx.x;
    const int lane = tid & 31;
    const int wv = tid >> 5;
    const int lhalf = lane & 15;
    const bool hi = lane >= 16;
    const int n0 = (tile * 8 + wv) << 4;

    const size_t base = (size_t)(b * NH + h) * N_PIX * HD;
    const unsigned long long kg = (unsigned long long)(uintptr_t)(Kb + base);
    const unsigned long long vg = (unsigned long long)(uintptr_t)(Vb + base);

    // Per-thread staging assignment: 256 16B chunks per tile, 1 each.
    const unsigned c16  = (unsigned)tid * 16u;
    const unsigned vgo0 = (unsigned)((tid >> 3) * 8192 + (tid & 7) * 16);

    // Q A-operand (constant for this wave)
    const bf16_t* qrow = Qb + base + (size_t)(n0 + lhalf) * HD;
    const v16bf aq = ld2(qrow + (hi ? 16 : 0), qrow + (hi ? 24 : 8));

    const float scale = 0.17677669529663687f;  // 1/sqrt(32)

    float mrow[8], lrow[8];
    v8f oa0, oa1, zc;
#pragma unroll
    for (int r = 0; r < 8; ++r) {
        mrow[r] = -3.0e38f; lrow[r] = 0.0f;
        oa0[r] = 0.0f; oa1[r] = 0.0f; zc[r] = 0.0f;
    }

    // --- async stage of chunk 0 into buffer 0 ---
    {
        const unsigned kl = ldsoff(&Kl[0][0][0]) + c16;
        const unsigned vl = ldsoff(&Vl[0][0][0]) + c16;
        asm volatile("global_load_async_to_lds_b128 %0, %1, %2"
                     :: "v"(kl), "v"(c16), "s"(kg) : "memory");
        asm volatile("global_load_async_to_lds_b128 %0, %1, %2"
                     :: "v"(vl), "v"(vgo0), "s"(vg) : "memory");
    }

    const int nch = N_PIX / MCH;
    for (int ci = 0; ci < nch; ++ci) {
        const int m0 = ci * MCH;
        const int buf = ci & 1;
        const bool hasnext = (ci + 1 < nch);

        if (hasnext) {   // prefetch chunk ci+1 into the other buffer
            const int nb = buf ^ 1;
            const unsigned kl = ldsoff(&Kl[nb][0][0]) + c16;
            const unsigned vl = ldsoff(&Vl[nb][0][0]) + c16;
            const unsigned kgo = (unsigned)((m0 + MCH) * (HD * 2)) + c16;
            const unsigned vgo = vgo0 + (unsigned)((m0 + MCH) * 2);
            asm volatile("global_load_async_to_lds_b128 %0, %1, %2"
                         :: "v"(kl), "v"(kgo), "s"(kg) : "memory");
            asm volatile("global_load_async_to_lds_b128 %0, %1, %2"
                         :: "v"(vl), "v"(vgo), "s"(vg) : "memory");
            asm volatile("s_wait_asynccnt 0x2" ::: "memory");
        } else {
            asm volatile("s_wait_asynccnt 0x0" ::: "memory");
        }
        __syncthreads();   // all waves' chunk-ci data in LDS

        // ---- scores: 4 WMMAs over 64 key columns ----
        v8f s[4];
#pragma unroll
        for (int j = 0; j < 4; ++j) {
            v16bf bk = *(const v16bf*)&Kl[buf][j * 16 + lhalf][hi ? 16 : 0];
            s[j] = wmma_bf16(aq, bk, zc);
        }

        // ---- online softmax over the 64-column chunk ----
#pragma unroll
        for (int r = 0; r < 8; ++r) {
            float a0 = s[0][r] * scale, a1 = s[1][r] * scale;
            float a2 = s[2][r] * scale, a3 = s[3][r] * scale;
            float cm = hmax16(fmaxf(fmaxf(a0, a1), fmaxf(a2, a3)));
            const float mn = fmaxf(mrow[r], cm);
            const float al = __expf(mrow[r] - mn);
            mrow[r] = mn;
            const float p0 = __expf(a0 - mn), p1 = __expf(a1 - mn);
            const float p2 = __expf(a2 - mn), p3 = __expf(a3 - mn);
            s[0][r] = p0; s[1][r] = p1; s[2][r] = p2; s[3][r] = p3;
            const float rs = hsum16((p0 + p1) + (p2 + p3));
            lrow[r] = lrow[r] * al + rs;
            oa0[r] *= al; oa1[r] *= al;
        }

        // ---- P -> LDS in mu-order: mu = (col%16)*4 + col/16  ----
        // lane's 4 tile-values for a row are contiguous: one ds_store_b64
#pragma unroll
        for (int r = 0; r < 8; ++r) {
            const int rr = r + (hi ? 8 : 0);
            v4bf pk;
            pk[0] = (bf16_t)s[0][r]; pk[1] = (bf16_t)s[1][r];
            pk[2] = (bf16_t)s[2][r]; pk[3] = (bf16_t)s[3][r];
            *(v4bf*)&pl[wv][rr][lhalf << 2] = pk;
        }
        asm volatile("s_wait_dscnt 0" ::: "memory");
        const bf16_t* prow = &pl[wv][lhalf][0];
        v16bf ap0 = ld2(prow + (hi ? 16 : 0),      prow + (hi ? 24 : 8));
        v16bf ap1 = ld2(prow + 32 + (hi ? 16 : 0), prow + 32 + (hi ? 24 : 8));

        // ---- P*V^T: 4 WMMAs (V already staged in mu-order) ----
        v16bf bv00 = *(const v16bf*)&Vl[buf][lhalf]     [(hi ? 16 : 0)];
        v16bf bv01 = *(const v16bf*)&Vl[buf][lhalf]     [32 + (hi ? 16 : 0)];
        v16bf bv10 = *(const v16bf*)&Vl[buf][16 + lhalf][(hi ? 16 : 0)];
        v16bf bv11 = *(const v16bf*)&Vl[buf][16 + lhalf][32 + (hi ? 16 : 0)];
        oa0 = wmma_bf16(ap0, bv00, oa0);
        oa0 = wmma_bf16(ap1, bv01, oa0);
        oa1 = wmma_bf16(ap0, bv10, oa1);
        oa1 = wmma_bf16(ap1, bv11, oa1);

        __syncthreads();   // everyone done with buf before it is re-staged
    }

    // normalize + store attn-out [b][h][n][d] bf16
#pragma unroll
    for (int r = 0; r < 8; ++r) {
        const float inv = 1.0f / lrow[r];
        const int n = n0 + r + (hi ? 8 : 0);
        AO[base + (size_t)n * HD + lhalf]      = (bf16_t)(oa0[r] * inv);
        AO[base + (size_t)n * HD + 16 + lhalf] = (bf16_t)(oa1[r] * inv);
    }
}

// ---------------------------------------------------------------------------
// Kernel 3: output projection + bias + residual (fp32 out).
// ---------------------------------------------------------------------------
__global__ __launch_bounds__(256) void proj_kernel(
    const bf16_t* __restrict__ AO, const bf16_t* __restrict__ wp,
    const float* __restrict__ bp, const float* __restrict__ x,
    float* __restrict__ out)
{
    const int blk = blockIdx.x;
    const int b  = blk >> 8;
    const int n0 = (blk & 255) << 4;
    const int tid = threadIdx.x;
    const int lane = tid & 31;
    const int wv = tid >> 5;
    const int lhalf = lane & 15;
    const bool hi = lane >= 16;

    v8f acc[2];
#pragma unroll
    for (int t = 0; t < 2; ++t)
#pragma unroll
        for (int i = 0; i < 8; ++i) acc[t][i] = 0.0f;

    for (int h = 0; h < NH; ++h) {
        const bf16_t* ab = AO + ((size_t)(b * NH + h) * N_PIX + n0 + lhalf) * HD
                              + (hi ? 8 : 0);
        v16bf bt = ld2(ab, ab + 16);
        const int cA0 = h * HD + (hi ? 8 : 0);
#pragma unroll
        for (int t = 0; t < 2; ++t) {
            const int o0 = (wv + 8 * t) << 4;
            const bf16_t* wrow = wp + (size_t)(o0 + lhalf) * C_IN;
            v16bf at = ld2(wrow + cA0, wrow + cA0 + 16);
            acc[t] = wmma_bf16(at, bt, acc[t]);
        }
    }

#pragma unroll
    for (int t = 0; t < 2; ++t) {
        const int o0 = (wv + 8 * t) << 4;
#pragma unroll
        for (int r = 0; r < 8; ++r) {
            const int M = r + (hi ? 8 : 0);
            const int o = o0 + M;
            const size_t idx = ((size_t)b * C_IN + o) * N_PIX + n0 + lhalf;
            out[idx] = acc[t][r] + bp[o] + x[idx];
        }
    }
}

// ---------------------------------------------------------------------------
// Host side
// ---------------------------------------------------------------------------
extern "C" void kernel_launch(void* const* d_in, const int* in_sizes, int n_in,
                              void* d_out, int out_size, void* d_ws, size_t ws_size,
                              hipStream_t stream) {
    const float* x      = (const float*)d_in[0];
    const float* w_qkv  = (const float*)d_in[1];
    const float* b_qkv  = (const float*)d_in[2];
    const float* w_proj = (const float*)d_in[3];
    const float* b_proj = (const float*)d_in[4];
    // d_in[5] = rel_bias: never applied in the reference forward.

    const size_t qkv_elems = (size_t)2 * NH * N_PIX * HD;  // 2,097,152
    bf16_t* Qb  = (bf16_t*)d_ws;
    bf16_t* Kb  = Qb + qkv_elems;
    bf16_t* Vb  = Kb + qkv_elems;
    bf16_t* AO  = Vb + qkv_elems;
    bf16_t* wqb = AO + qkv_elems;          // 768*256 bf16
    bf16_t* wpb = wqb + 768 * C_IN;        // 256*256 bf16

    float* out = (float*)d_out;

    wconv_kernel<<<768, 256, 0, stream>>>(w_qkv, w_proj, wqb, wpb);
    qkv_kernel  <<<512, 256, 0, stream>>>(x, wqb, b_qkv, Qb, Kb, Vb);
    attn_kernel <<<512, 256, 0, stream>>>(Qb, Kb, Vb, AO);
    proj_kernel <<<512, 256, 0, stream>>>(AO, wpb, b_proj, x, out);
}
